// TransformerBlock_2774548873275
// MI455X (gfx1250) — compile-verified, matrix operand order checked by reading, not run on hardware
//
#include <hip/hip_runtime.h>
#include <hip/hip_bf16.h>

// ---------------------------------------------------------------------------
// MI455X (gfx1250) transformer block, bf16 WMMA everywhere, fp32 accumulate.
// GEMMs use GLOBAL_LOAD_ASYNC_TO_LDS_B128 (ASYNCcnt) with double-buffered LDS
// so tile k+1 streams in while tile k feeds the WMMA pipes.
// Two-pass "flash" attention adapted to the reference's softmax-over-query
// convention: pass1 = per-key logsumexp, pass2 = S^T recompute + P*V.
// ---------------------------------------------------------------------------

typedef __attribute__((ext_vector_type(16))) __bf16 bf16x16;
typedef __attribute__((ext_vector_type(8)))  float  f32x8;

union Frag16 { bf16x16 v; uint4 q[2]; };
union Pack16 { bf16x16 v; unsigned short s[16]; };

__device__ __forceinline__ unsigned short f2bf(float f) {
  unsigned u = __float_as_uint(f);
  unsigned r = (u + 0x7FFFu + ((u >> 16) & 1u)) >> 16;   // round-nearest-even
  return (unsigned short)r;
}

__device__ __forceinline__ f32x8 wmma_bf16(bf16x16 a, bf16x16 b, f32x8 c) {
  // v_wmma_f32_16x16x32_bf16: D = A(16x32) * B(32x16) + C(16x16 f32)
  return __builtin_amdgcn_wmma_f32_16x16x32_bf16(
      false, a, false, b, (short)0, c, false, false);
}

// Async copy 16B global -> LDS (per lane), tracked by ASYNCcnt.  The LDS
// destination VGPR holds the workgroup-relative LDS byte offset, which is the
// low 32 bits of the flat shared-memory address (LDS aperture, ISA 10.2).
__device__ __forceinline__ void async_ld_b128(void* lds_ptr, const void* gptr) {
  unsigned off = (unsigned)(size_t)lds_ptr;
  asm volatile("global_load_async_to_lds_b128 %0, %1, off"
               :: "v"(off), "v"(gptr)
               : "memory");
}

__device__ __forceinline__ void wait_async0() {
  asm volatile("s_wait_asynccnt 0x0" ::: "memory");
}

// ---------------------------------------------------------------------------
// fp32 -> bf16 converter (grid-stride)
// ---------------------------------------------------------------------------
__global__ void cvt_f32_bf16(const float* __restrict__ in,
                             unsigned short* __restrict__ out, int n) {
  for (int i = blockIdx.x * blockDim.x + threadIdx.x; i < n;
       i += gridDim.x * blockDim.x)
    out[i] = f2bf(in[i]);
}

// ---------------------------------------------------------------------------
// Generic bf16 GEMM: C[M,N] = A[M,K] * B[N,K]^T, fp32 accumulation.
// 128x128 workgroup tile, 8 waves (wave32) each doing 2x4 16x16 WMMA tiles,
// K-step 32.  LDS tiles padded to 40 elements/row (80B, 16B aligned),
// double-buffered and filled by async-to-LDS loads (no VGPR staging).
// Epilogue modes:
//   0: scatter bf16 -> [b,a,c,h]          (Q/K projections)
//   1: scatter bf16 -> [b,a,h,c]          (V projection, transposed)
//   2: f32 out = acc + resid; bf16 copy    (W_O: xr = x + attn_out)
//   3: bf16 out = relu(acc + bias[n])      (W_in)
//   4: f32 out = acc + bias[n] + resid     (W_out final)
// ---------------------------------------------------------------------------
__launch_bounds__(256)
__global__ void gemm_bf16k(const unsigned short* __restrict__ A,
                           const unsigned short* __restrict__ Bm,
                           int M, int N, int K,
                           float* __restrict__ outf,
                           unsigned short* __restrict__ outb,
                           const float* __restrict__ bias,
                           const float* __restrict__ resid,
                           int mode) {
  __shared__ __align__(16) unsigned short sA[2][128 * 40];
  __shared__ __align__(16) unsigned short sB[2][128 * 40];

  const int tid  = threadIdx.x;
  const int lane = tid & 31;
  const int wave = tid >> 5;
  const int wm   = wave & 3;           // 4 waves down M (32 rows each)
  const int wn   = wave >> 2;          // 2 waves across N (64 cols each)
  const int l    = lane & 15;
  const int koff = (lane < 16) ? 0 : 8;
  const int m0   = blockIdx.y * 128;
  const int n0   = blockIdx.x * 128;

  // This thread's slice of the cooperative tile copy: 2 x 16B per matrix.
  const int r0  = tid >> 2,        ch0 = (tid & 3) * 8;
  const int r1  = (tid + 256) >> 2, ch1 = ((tid + 256) & 3) * 8;

  f32x8 acc[2][4] = {};

  // Prologue: stream tile k0=0 into buffer 0.
  async_ld_b128(&sA[0][r0 * 40 + ch0], A  + (size_t)(m0 + r0) * K + ch0);
  async_ld_b128(&sB[0][r0 * 40 + ch0], Bm + (size_t)(n0 + r0) * K + ch0);
  async_ld_b128(&sA[0][r1 * 40 + ch1], A  + (size_t)(m0 + r1) * K + ch1);
  async_ld_b128(&sB[0][r1 * 40 + ch1], Bm + (size_t)(n0 + r1) * K + ch1);
  wait_async0();
  __syncthreads();

  for (int k0 = 0; k0 < K; k0 += 32) {
    const int cur = (k0 >> 5) & 1;
    const int nxt = cur ^ 1;
    if (k0 + 32 < K) {  // prefetch next K-tile while WMMAs run on this one
      const int kn = k0 + 32;
      async_ld_b128(&sA[nxt][r0 * 40 + ch0], A  + (size_t)(m0 + r0) * K + kn + ch0);
      async_ld_b128(&sB[nxt][r0 * 40 + ch0], Bm + (size_t)(n0 + r0) * K + kn + ch0);
      async_ld_b128(&sA[nxt][r1 * 40 + ch1], A  + (size_t)(m0 + r1) * K + kn + ch1);
      async_ld_b128(&sB[nxt][r1 * 40 + ch1], Bm + (size_t)(n0 + r1) * K + kn + ch1);
    }

    Frag16 af[2], bf[4];
#pragma unroll
    for (int mi = 0; mi < 2; ++mi) {
      const unsigned short* p = &sA[cur][(wm * 32 + mi * 16 + l) * 40 + koff];
      af[mi].q[0] = *(const uint4*)(p);
      af[mi].q[1] = *(const uint4*)(p + 16);
    }
#pragma unroll
    for (int ni = 0; ni < 4; ++ni) {
      const unsigned short* p = &sB[cur][(wn * 64 + ni * 16 + l) * 40 + koff];
      bf[ni].q[0] = *(const uint4*)(p);
      bf[ni].q[1] = *(const uint4*)(p + 16);
    }
#pragma unroll
    for (int mi = 0; mi < 2; ++mi)
#pragma unroll
      for (int ni = 0; ni < 4; ++ni)
        acc[mi][ni] = wmma_bf16(af[mi].v, bf[ni].v, acc[mi][ni]);

    wait_async0();     // my prefetch writes to buf[nxt] have landed in LDS
    __syncthreads();   // publishes everyone's; dscnt waited before signal
  }

  const int off8 = (lane < 16) ? 0 : 8;
#pragma unroll
  for (int mi = 0; mi < 2; ++mi) {
#pragma unroll
    for (int ni = 0; ni < 4; ++ni) {
#pragma unroll
      for (int r = 0; r < 8; ++r) {
        int gm = m0 + wm * 32 + mi * 16 + off8 + r;
        int gn = n0 + wn * 64 + ni * 16 + l;
        float v = acc[mi][ni][r];
        if (mode <= 1) {
          int b = gm >> 11, c = gm & 2047;      // SEQ = 2048
          int hd = gn >> 6, h = gn & 63;        // D_HEAD = 64
          size_t idx = (mode == 0)
              ? ((((size_t)(b * 16 + hd)) * 2048 + c) * 64 + h)
              : ((((size_t)(b * 16 + hd)) * 64 + h) * 2048 + c);
          outb[idx] = f2bf(v);
        } else if (mode == 2) {
          size_t idx = (size_t)gm * N + gn;
          float o = v + resid[idx];
          outf[idx] = o;
          outb[idx] = f2bf(o);
        } else if (mode == 3) {
          size_t idx = (size_t)gm * N + gn;
          float o = fmaxf(v + bias[gn], 0.0f);
          outb[idx] = f2bf(o);
        } else {
          size_t idx = (size_t)gm * N + gn;
          outf[idx] = v + bias[gn] + resid[idx];
        }
      }
    }
  }
}

// ---------------------------------------------------------------------------
// Attention pass 1: per-key logsumexp.
//   d_c = m_c + log( sum_{C<=c} exp( (k_c . q_C) / 8 - m_c ) )
// Each wave owns 16 key rows (B operand, N=c); loops query tiles (A, M=C).
// D-tile: lane = c (lanes l and l+16 share c, split over C rows 0-7 / 8-15);
// halves combined with shfl_xor(16) at the end (wave32).
// ---------------------------------------------------------------------------
__launch_bounds__(256)
__global__ void attn_pass1(const unsigned short* __restrict__ qb,
                           const unsigned short* __restrict__ kb,
                           float* __restrict__ dstats) {
  const int bh   = blockIdx.y;                 // b*16 + head
  const int blk  = blockIdx.x;                 // 128-key slab
  const int wave = threadIdx.x >> 5;
  const int lane = threadIdx.x & 31;
  const int l    = lane & 15;
  const int koff = (lane < 16) ? 0 : 8;
  const int Mb   = (lane < 16) ? 0 : 8;
  const int c0w  = blk * 128 + wave * 16;
  const size_t base = (size_t)bh * 2048 * 64;

  Frag16 kf0, kf1;  // B operand: K rows c0w..c0w+15 (h contiguous)
  {
    const unsigned short* kr = kb + base + (size_t)(c0w + l) * 64;
    kf0.q[0] = *(const uint4*)(kr + koff);
    kf0.q[1] = *(const uint4*)(kr + koff + 16);
    kf1.q[0] = *(const uint4*)(kr + 32 + koff);
    kf1.q[1] = *(const uint4*)(kr + 32 + koff + 16);
  }

  float m = -1e30f, lsum = 0.0f;
  const int cL = c0w + l;

  for (int C0 = 0; C0 < c0w + 16; C0 += 16) {
    Frag16 qf0, qf1;  // A operand: Q rows C0..C0+15
    const unsigned short* qr = qb + base + (size_t)(C0 + l) * 64;
    qf0.q[0] = *(const uint4*)(qr + koff);
    qf0.q[1] = *(const uint4*)(qr + koff + 16);
    qf1.q[0] = *(const uint4*)(qr + 32 + koff);
    qf1.q[1] = *(const uint4*)(qr + 32 + koff + 16);

    f32x8 s = {};
    s = wmma_bf16(qf0.v, kf0.v, s);
    s = wmma_bf16(qf1.v, kf1.v, s);

    float t[8], tmax = -1e30f;
#pragma unroll
    for (int r = 0; r < 8; ++r) {
      int C = C0 + Mb + r;
      t[r] = (C <= cL) ? s[r] * 0.125f : -1.25e9f;  // (-1e10)/sqrt(64)
      tmax = fmaxf(tmax, t[r]);
    }
    float mn = fmaxf(m, tmax);
    float add = 0.0f;
#pragma unroll
    for (int r = 0; r < 8; ++r) add += __expf(t[r] - mn);
    lsum = lsum * __expf(m - mn) + add;
    m = mn;
  }

  float mo = __shfl_xor(m, 16, 32);
  float lo = __shfl_xor(lsum, 16, 32);
  float mn = fmaxf(m, mo);
  float ltot = lsum * __expf(m - mn) + lo * __expf(mo - mn);
  if (lane < 16) dstats[(size_t)bh * 2048 + cL] = mn + __logf(ltot);
}

// ---------------------------------------------------------------------------
// Attention pass 2: Z[C,h] = sum_{c>=C} exp(S[c,C]/8 - d_c) * V[c,h].
// S computed as D[M=c, N=C] (A = K rows, B = Q rows).  Two consecutive
// c-tiles' D layout coincides with the bf16 A-operand layout of P[C, K=c],
// so P feeds the Z WMMAs with no LDS transpose.  V pre-transposed [b,a,h,c].
// ---------------------------------------------------------------------------
__launch_bounds__(256)
__global__ void attn_pass2(const unsigned short* __restrict__ qb,
                           const unsigned short* __restrict__ kb,
                           const unsigned short* __restrict__ vt,
                           const float* __restrict__ dstats,
                           unsigned short* __restrict__ zf) {
  const int bh   = blockIdx.y;
  const int b    = bh >> 4, head = bh & 15;
  const int blk  = blockIdx.x;
  const int wave = threadIdx.x >> 5;
  const int lane = threadIdx.x & 31;
  const int l    = lane & 15;
  const int koff = (lane < 16) ? 0 : 8;
  const int off8 = (lane < 16) ? 0 : 8;
  const int C0w  = blk * 128 + wave * 16;
  const size_t base = (size_t)bh * 2048 * 64;

  Frag16 qf0, qf1;  // B operand (fixed): Q rows C0w..C0w+15
  {
    const unsigned short* qr = qb + base + (size_t)(C0w + l) * 64;
    qf0.q[0] = *(const uint4*)(qr + koff);
    qf0.q[1] = *(const uint4*)(qr + koff + 16);
    qf1.q[0] = *(const uint4*)(qr + 32 + koff);
    qf1.q[1] = *(const uint4*)(qr + 32 + koff + 16);
  }

  f32x8 zacc[4] = {};
  const int CL = C0w + l;                       // this lane's query index
  const float* drow = dstats + (size_t)bh * 2048;

  const int jstart = (blk * 8 + wave) & ~1;     // pair-aligned key tile
  for (int j = jstart; j < 128; j += 2) {
    const int cb = j * 16;                      // 32 keys cb..cb+31
    Frag16 kf0, kf1, kg0, kg1;                  // A operands: K rows
    const unsigned short* kr = kb + base + (size_t)(cb + l) * 64;
    kf0.q[0] = *(const uint4*)(kr + koff);
    kf0.q[1] = *(const uint4*)(kr + koff + 16);
    kf1.q[0] = *(const uint4*)(kr + 32 + koff);
    kf1.q[1] = *(const uint4*)(kr + 32 + koff + 16);
    const unsigned short* kr2 = kr + 16 * 64;
    kg0.q[0] = *(const uint4*)(kr2 + koff);
    kg0.q[1] = *(const uint4*)(kr2 + koff + 16);
    kg1.q[0] = *(const uint4*)(kr2 + 32 + koff);
    kg1.q[1] = *(const uint4*)(kr2 + 32 + koff + 16);

    f32x8 s0 = {}, s1 = {};
    s0 = wmma_bf16(kf0.v, qf0.v, s0);
    s0 = wmma_bf16(kf1.v, qf1.v, s0);
    s1 = wmma_bf16(kg0.v, qf0.v, s1);
    s1 = wmma_bf16(kg1.v, qf1.v, s1);

    Pack16 p;   // becomes A operand P[C=M, K=c(32)] directly
#pragma unroll
    for (int r = 0; r < 8; ++r) {
      int c = cb + off8 + r;
      float val = (CL <= c) ? __expf(s0[r] * 0.125f - drow[c]) : 0.0f;
      p.s[r] = f2bf(val);
      int c2 = c + 16;
      float val2 = (CL <= c2) ? __expf(s1[r] * 0.125f - drow[c2]) : 0.0f;
      p.s[8 + r] = f2bf(val2);
    }

#pragma unroll
    for (int ht = 0; ht < 4; ++ht) {            // 4 h-tiles cover D_HEAD=64
      Frag16 vf;  // B operand: V^T rows h, c contiguous
      const unsigned short* vr =
          vt + ((size_t)bh * 64 + ht * 16 + l) * 2048 + cb + koff;
      vf.q[0] = *(const uint4*)(vr);
      vf.q[1] = *(const uint4*)(vr + 16);
      zacc[ht] = wmma_bf16(p.v, vf.v, zacc[ht]);
    }
  }

  // z_flat[b, C, head*64 + h] (bf16)
#pragma unroll
  for (int ht = 0; ht < 4; ++ht) {
#pragma unroll
    for (int r = 0; r < 8; ++r) {
      int C = C0w + off8 + r;
      int col = head * 64 + ht * 16 + l;
      zf[((size_t)b * 2048 + C) * 1024 + col] = f2bf(zacc[ht][r]);
    }
  }
}

// ---------------------------------------------------------------------------
// Host orchestration
// ---------------------------------------------------------------------------
extern "C" void kernel_launch(void* const* d_in, const int* in_sizes, int n_in,
                              void* d_out, int out_size, void* d_ws,
                              size_t ws_size, hipStream_t stream) {
  (void)in_sizes; (void)n_in; (void)out_size; (void)ws_size;
  const float* x     = (const float*)d_in[0];
  const float* W_K   = (const float*)d_in[1];
  const float* W_Q   = (const float*)d_in[2];
  const float* W_V   = (const float*)d_in[3];
  const float* W_O   = (const float*)d_in[4];
  const float* W_in  = (const float*)d_in[5];
  const float* b_in  = (const float*)d_in[6];
  const float* W_out = (const float*)d_in[7];
  const float* b_out = (const float*)d_in[8];
  float* out = (float*)d_out;

  const int Mrows = 4096;  // BATCH * SEQ
  size_t off = 0;
  char* wsb = (char*)d_ws;
  auto alloc = [&](size_t bytes) -> char* {
    char* p = wsb + off;
    off += (bytes + 255) & ~(size_t)255;
    return p;
  };
  unsigned short* xb    = (unsigned short*)alloc((size_t)Mrows * 1024 * 2);
  unsigned short* wqb   = (unsigned short*)alloc((size_t)1024 * 1024 * 2);
  unsigned short* wkb   = (unsigned short*)alloc((size_t)1024 * 1024 * 2);
  unsigned short* wvb   = (unsigned short*)alloc((size_t)1024 * 1024 * 2);
  unsigned short* wob   = (unsigned short*)alloc((size_t)1024 * 1024 * 2);
  unsigned short* winb  = (unsigned short*)alloc((size_t)4096 * 1024 * 2);
  unsigned short* woutb = (unsigned short*)alloc((size_t)1024 * 4096 * 2);
  unsigned short* qbuf  = (unsigned short*)alloc((size_t)Mrows * 1024 * 2);
  unsigned short* kbuf  = (unsigned short*)alloc((size_t)Mrows * 1024 * 2);
  unsigned short* vtb   = (unsigned short*)alloc((size_t)Mrows * 1024 * 2);
  float*          dst   = (float*)alloc((size_t)32 * 2048 * 4);
  unsigned short* zf    = (unsigned short*)alloc((size_t)Mrows * 1024 * 2);
  unsigned short* xrb   = (unsigned short*)alloc((size_t)Mrows * 1024 * 2);
  unsigned short* hb    = (unsigned short*)alloc((size_t)Mrows * 4096 * 2);

  const dim3 blk(256);
  cvt_f32_bf16<<<1024, blk, 0, stream>>>(x,     xb,    Mrows * 1024);
  cvt_f32_bf16<<<1024, blk, 0, stream>>>(W_Q,   wqb,   1024 * 1024);
  cvt_f32_bf16<<<1024, blk, 0, stream>>>(W_K,   wkb,   1024 * 1024);
  cvt_f32_bf16<<<1024, blk, 0, stream>>>(W_V,   wvb,   1024 * 1024);
  cvt_f32_bf16<<<1024, blk, 0, stream>>>(W_O,   wob,   1024 * 1024);
  cvt_f32_bf16<<<1024, blk, 0, stream>>>(W_in,  winb,  4096 * 1024);
  cvt_f32_bf16<<<1024, blk, 0, stream>>>(W_out, woutb, 1024 * 4096);

  const dim3 g8(8, 32);    // N=1024 tiles x M=4096 tiles
  const dim3 g32(32, 32);  // N=4096
  // Q/K -> [b,a,c,h]; V -> [b,a,h,c] (transposed for contiguous B-frag loads)
  gemm_bf16k<<<g8, blk, 0, stream>>>(xb, wqb, Mrows, 1024, 1024,
                                     nullptr, qbuf, nullptr, nullptr, 0);
  gemm_bf16k<<<g8, blk, 0, stream>>>(xb, wkb, Mrows, 1024, 1024,
                                     nullptr, kbuf, nullptr, nullptr, 0);
  gemm_bf16k<<<g8, blk, 0, stream>>>(xb, wvb, Mrows, 1024, 1024,
                                     nullptr, vtb, nullptr, nullptr, 1);

  const dim3 ag(16, 32);   // 16 slabs of 128 rows x (B*H = 32)
  attn_pass1<<<ag, blk, 0, stream>>>(qbuf, kbuf, dst);
  attn_pass2<<<ag, blk, 0, stream>>>(qbuf, kbuf, vtb, dst, zf);

  // xr = x + z_flat * W_O^T   (fp32 into d_out, bf16 copy for MLP)
  gemm_bf16k<<<g8, blk, 0, stream>>>(zf, wob, Mrows, 1024, 1024,
                                     out, xrb, nullptr, x, 2);
  // h = relu(xr * W_in^T + b_in)
  gemm_bf16k<<<g32, blk, 0, stream>>>(xrb, winb, Mrows, 4096, 1024,
                                      nullptr, hb, b_in, nullptr, 3);
  // out = xr + h * W_out^T + b_out
  gemm_bf16k<<<g8, blk, 0, stream>>>(hb, woutb, Mrows, 1024, 4096,
                                     out, nullptr, b_out, out, 4);
}
